// Scaled_Dot_Product_Attention_31421980738326
// MI455X (gfx1250) — matrix operand order repaired and checked by reading, not compiled, not run on hardware
//
#include <hip/hip_runtime.h>
#include <hip/hip_bf16.h>

typedef __attribute__((ext_vector_type(16))) _Float16 v16h;
typedef __attribute__((ext_vector_type(2)))  __fp16   f16x2;
typedef __attribute__((ext_vector_type(8)))  float    v8f;
typedef __attribute__((ext_vector_type(4)))  unsigned int v4u;
typedef __attribute__((ext_vector_type(8)))  int      v8i;
typedef __attribute__((ext_vector_type(4)))  int      v4i;

#define B_   16
#define L_   2048
#define D_   128
#define NEGV (-1.0e9f)

#define QTILE 128       // query rows per block (8 waves x 16)
#define KT    32        // keys per iteration
#define KRSTR 130       // raw K tile row stride in floats (TDM pad: 128 dw + 2 dw)
#define KSTR  136       // packed Klds row stride (f16), padded vs 128
#define VSTR  40        // Vlds row stride (f16), padded vs 32
#define PSTR  40        // P buffer row stride (f16)

union HU2   { f16x2 h; uint32_t u; };
union V16HU { v16h v; uint32_t u[8]; };

__device__ inline uint32_t pack2(float a, float b) {
    HU2 x; x.h = __builtin_amdgcn_cvt_pkrtz(a, b); return x.u;
}

// Issue a TDM load of one K tile: 32 rows x 128 fp32, row-major, into LDS at
// lds_off with a 2-DWORD pad after every 128-DWORD row (row stride 520 B).
__device__ inline void tdm_load_k_tile(const float* gptr, uint32_t lds_off) {
    uint64_t ga = (uint64_t)(uintptr_t)gptr;
    v4u g0;
    g0[0] = 1u;                                            // count=1, user load
    g0[1] = lds_off;                                       // lds_addr (bytes)
    g0[2] = (uint32_t)ga;                                  // global_addr[31:0]
    g0[3] = (uint32_t)((ga >> 32) & 0x01ffffffu)           // global_addr[56:32]
          | (2u << 30);                                    // type = 2 ("image")
    v8i g1;
    g1[0] = (int)((2u << 16)        // data_size = 4 bytes
                | (1u << 20)        // pad_enable
                | (6u << 22)        // pad_interval: 128 DWORDs
                | (1u << 25));      // pad_amount: 2 DWORDs
    g1[1] = (int)(128u << 16);      // tensor_dim0 = 128 (low16 at bits [63:48])
    g1[2] = (int)(2048u << 16);     // tensor_dim1 = 2048 (low16 at bits [95:80])
    g1[3] = (int)(128u << 16);      // tile_dim0 = 128 elements
    g1[4] = (int)32;                // tile_dim1 = 32 rows (tile_dim2 = 0)
    g1[5] = (int)128;               // tensor_dim0_stride = 128 elements
    g1[6] = 0;
    g1[7] = 0;
    v4i z4 = (v4i){0, 0, 0, 0};     // groups 2/3 unused (2-D tensor)
    v8i z8 = (v8i){0, 0, 0, 0, 0, 0, 0, 0};
    __builtin_amdgcn_tensor_load_to_lds(g0, g1, z4, z4, z8, 0);
}

__global__ __launch_bounds__(256)
void fa_wmma_kernel(const float* __restrict__ Q, const float* __restrict__ K,
                    const float* __restrict__ V, const float* __restrict__ scale_p,
                    const int* __restrict__ mask, float* __restrict__ out)
{
    __shared__ __align__(16) float    Kraw[2][KT * KRSTR];   // 2 x 16640 B (TDM dest)
    __shared__ __align__(16) _Float16 Klds[KT * KSTR];       //  8704 B (packed f16)
    __shared__ __align__(16) _Float16 Vlds[D_ * VSTR];       // 10240 B (transposed [d][key])
    __shared__ __align__(16) _Float16 Plds[8 * 16 * PSTR];   // 10240 B (per-wave P bounce)

    const int tid  = threadIdx.x;
    const int lane = tid & 31;
    const int wave = tid >> 5;
    const int hi   = (lane >> 4) & 1;
    const int ln   = lane & 15;

    const int blocksPerB = L_ / QTILE;
    const int b  = blockIdx.x / blocksPerB;
    const int qt = blockIdx.x % blocksPerB;
    const int q0 = qt * QTILE + wave * 16;

    const float scale = scale_p[0];
    const float* Kbase = K + (size_t)b * L_ * D_;
    const float* Vbase = V + (size_t)b * L_ * D_;

    // ---- Q -> A fragments (16x32 f16 layout), 4 chunks over D=128 ----
    v16h qf[4];
    {
        const float* qrow = Q + ((size_t)b * L_ + (size_t)(q0 + ln)) * D_;
        #pragma unroll
        for (int c = 0; c < 4; ++c) {
            V16HU f;
            #pragma unroll
            for (int v = 0; v < 8; ++v) {
                int kd = 32*c + ((v < 4) ? (2*v + 8*hi) : (16 + 2*(v-4) + 8*hi));
                float2 d2 = *(const float2*)(qrow + kd);
                f.u[v] = pack2(d2.x, d2.y);
            }
            qf[c] = f.v;
        }
    }

    // ---- per-row mask flags (row = v + 8*hi) ----
    bool rowmask[8];
    #pragma unroll
    for (int v = 0; v < 8; ++v)
        rowmask[v] = (mask[b * L_ + q0 + v + 8*hi] == -1);

    v8f acc[8];
    #pragma unroll
    for (int t = 0; t < 8; ++t) acc[t] = (v8f){0,0,0,0,0,0,0,0};
    float mrow[8], lrow[8];
    #pragma unroll
    for (int v = 0; v < 8; ++v) { mrow[v] = -INFINITY; lrow[v] = 0.0f; }

    _Float16* Pw = Plds + wave * (16 * PSTR);

    // ---- TDM prologue: prefetch K tile 0 into buffer 0 ----
    if (wave == 0)
        tdm_load_k_tile(Kbase, (uint32_t)(uintptr_t)&Kraw[0][0]);

    const int ntiles = L_ / KT;
    for (int tile = 0; tile < ntiles; ++tile) {
        const int j0  = tile * KT;
        const int buf = tile & 1;
        __syncthreads();   // previous tile fully consumed; other raw buffer free

        // ---- stage V tile transposed: Vlds[d][key] (f16) ----
        {
            const float* Vg = Vbase + (size_t)j0 * D_;
            #pragma unroll
            for (int i = 0; i < 4; ++i) {
                int task = tid + i * 256;       // 0..1023
                int d  = task & 127;
                int k4 = (task >> 7) << 2;      // 0,4,...,28
                float x0 = Vg[(size_t)(k4 + 0) * D_ + d];
                float x1 = Vg[(size_t)(k4 + 1) * D_ + d];
                float x2 = Vg[(size_t)(k4 + 2) * D_ + d];
                float x3 = Vg[(size_t)(k4 + 3) * D_ + d];
                uint2 p;
                p.x = pack2(x0, x1);
                p.y = pack2(x2, x3);
                *(uint2*)(&Vlds[d * VSTR + k4]) = p;
            }
            if (tile + 1 < ntiles)   // warm L2/WGP$ for next V tile
                __builtin_prefetch(Vg + (size_t)KT * D_ + tid * 16, 0, 1);
        }

        // ---- TDM: issue next K tile, wait for current one ----
        if (wave == 0) {
            if (tile + 1 < ntiles) {
                tdm_load_k_tile(Kbase + (size_t)(j0 + KT) * D_,
                                (uint32_t)(uintptr_t)&Kraw[buf ^ 1][0]);
                __builtin_amdgcn_s_wait_tensorcnt(1);   // oldest (current tile) done
            } else {
                __builtin_amdgcn_s_wait_tensorcnt(0);
            }
        }
        __syncthreads();   // raw K tile in LDS + V staged, visible to all waves

        // ---- cooperative repack: raw fp32 K tile -> packed f16 Klds ----
        {
            const float* kr = &Kraw[buf][0];
            #pragma unroll
            for (int i = 0; i < 4; ++i) {
                int task = tid + i * 256;       // 0..1023, 4 f16 each
                int key  = task >> 5;
                int d4   = (task & 31) << 2;
                const float* src = kr + key * KRSTR + d4;
                float2 a = *(const float2*)(src + 0);
                float2 c = *(const float2*)(src + 2);
                uint2 p;
                p.x = pack2(a.x, a.y);
                p.y = pack2(c.x, c.y);
                *(uint2*)(&Klds[key * KSTR + d4]) = p;
            }
        }
        __syncthreads();   // packed Klds ready

        // ---- S = Q * K^T : B fragments from packed f16 Klds ----
        v8f s[2];
        #pragma unroll
        for (int half = 0; half < 2; ++half) {
            v8f sacc = (v8f){0,0,0,0,0,0,0,0};
            const _Float16* krow = &Klds[(16*half + ln) * KSTR];
            #pragma unroll
            for (int c = 0; c < 4; ++c) {
                V16HU bf;
                #pragma unroll
                for (int v = 0; v < 8; ++v) {
                    int kd = 32*c + ((v < 4) ? (2*v + 8*hi) : (16 + 2*(v-4) + 8*hi));    // B 32x16 layout: k pairs per vgpr, table order
                    bf.u[v] = *(const uint32_t*)(krow + kd);
                }
                sacc = __builtin_amdgcn_wmma_f32_16x16x32_f16(
                           false, qf[c], false, bf.v, (short)0, sacc, false, false);
            }
            s[half] = sacc;
        }

        // ---- scale + row mask ----
        #pragma unroll
        for (int v = 0; v < 8; ++v) {
            float a0 = s[0][v] * scale;
            float a1 = s[1][v] * scale;
            if (rowmask[v]) { a0 = NEGV; a1 = NEGV; }
            s[0][v] = a0; s[1][v] = a1;
        }

        // ---- online softmax (row = v + 8*hi; reduce over 16 lanes of half) ----
        float corr[8], p0[8], p1[8];
        #pragma unroll
        for (int v = 0; v < 8; ++v) {
            float m = fmaxf(s[0][v], s[1][v]);
            #pragma unroll
            for (int off = 8; off >= 1; off >>= 1)
                m = fmaxf(m, __shfl_xor(m, off, 32));
            float mnew = fmaxf(mrow[v], m);
            corr[v] = __expf(mrow[v] - mnew);
            p0[v]   = __expf(s[0][v] - mnew);
            p1[v]   = __expf(s[1][v] - mnew);
            float su = p0[v] + p1[v];
            #pragma unroll
            for (int off = 8; off >= 1; off >>= 1)
                su += __shfl_xor(su, off, 32);
            lrow[v] = lrow[v] * corr[v] + su;
            mrow[v] = mnew;
        }
        #pragma unroll
        for (int t = 0; t < 8; ++t)
            #pragma unroll
            for (int v = 0; v < 8; ++v)
                acc[t][v] *= corr[v];

        // ---- bounce P through LDS: C/D layout -> A layout ----
        #pragma unroll
        for (int v = 0; v < 8; ++v) {
            int row = v + 8*hi;
            Pw[row * PSTR + ln]      = (_Float16)p0[v];
            Pw[row * PSTR + 16 + ln] = (_Float16)p1[v];
        }
        asm volatile("" ::: "memory");   // same-wave DS ops stay in order

        v16h pf;
        {
            V16HU f;
            #pragma unroll
            for (int v = 0; v < 8; ++v) {
                int k = (v < 4) ? (2*v + 8*hi) : (16 + 2*(v-4) + 8*hi);
                f.u[v] = *(const uint32_t*)(&Pw[ln * PSTR + k]);
            }
            pf = f.v;
        }

        // ---- O += P * V : 8 d-tiles, K=32 each ----
        #pragma unroll
        for (int t = 0; t < 8; ++t) {
            V16HU bf;
            #pragma unroll
            for (int v = 0; v < 8; ++v) {
                int k = (v < 4) ? (2*v + 8*hi) : (16 + 2*(v-4) + 8*hi);
                bf.u[v] = *(const uint32_t*)(&Vlds[(16*t + ln) * VSTR + k]);
            }
            acc[t] = __builtin_amdgcn_wmma_f32_16x16x32_f16(
                         false, pf, false, bf.v, (short)0, acc[t], false, false);
        }
    }

    // ---- normalize + store ----
    #pragma unroll
    for (int t = 0; t < 8; ++t) {
        int d = 16*t + ln;
        #pragma unroll
        for (int v = 0; v < 8; ++v) {
            int q = q0 + v + 8*hi;
            out[((size_t)b * L_ + q) * D_ + d] = acc[t][v] / lrow[v];
        }
    }
}

extern "C" void kernel_launch(void* const* d_in, const int* in_sizes, int n_in,
                              void* d_out, int out_size, void* d_ws, size_t ws_size,
                              hipStream_t stream) {
    const float* Q     = (const float*)d_in[0];
    const float* K     = (const float*)d_in[1];
    const float* V     = (const float*)d_in[2];
    const float* scale = (const float*)d_in[3];
    const int*   mask  = (const int*)d_in[4];
    float*       out   = (float*)d_out;

    dim3 grid(B_ * (L_ / QTILE));
    dim3 block(256);
    fa_wmma_kernel<<<grid, block, 0, stream>>>(Q, K, V, scale, mask, out);
}
